// MPNN_16466904613491
// MI455X (gfx1250) — compile-verified, matrix operand order checked by reading, not run on hardware
//
#include <hip/hip_runtime.h>

#define DM     128
#define GPB    2            // graphs per block
#define NPG    10
#define EPG    50
#define NODES  (GPB * NPG)  // 20
#define EDGES  (GPB * EPG)  // 100
#define ETILES 7            // ceil(EDGES/16)
#define NTILES 2            // ceil(NODES/16)
#define KSTEPS 8            // 256 / 32

typedef __attribute__((ext_vector_type(16))) _Float16 v16h;
typedef __attribute__((ext_vector_type(8)))  float    v8f;
typedef int vi4 __attribute__((vector_size(16)));   // param type of async b128 builtin

// Async global->LDS path (gfx1250 GLOBAL_LOAD_ASYNC_TO_LDS_*, ASYNCcnt-tracked).
// Guarded: falls back to regular loads if the toolchain doesn't declare the builtins.
#if defined(__gfx1250__) &&                                                  \
    __has_builtin(__builtin_amdgcn_global_load_async_to_lds_b128) &&         \
    __has_builtin(__builtin_amdgcn_s_wait_asynccnt)
#define HAVE_ASYNC_LDS 1
#else
#define HAVE_ASYNC_LDS 0
#endif

struct MpnnParams {
    const float* x;        // [N,16]
    const int*   ei;       // [2,E]
    const float* ea;       // [E,16]
    const float* y;        // [B,5]
    const float* ne_w; const float* ne_b;   // [16,128],[128]
    const float* ee_w; const float* ee_b;   // [16,128],[128]
    const float* msg_w[4]; const float* msg_b[4];  // [256,128],[128]
    const float* upd_w[4]; const float* upd_b[4];  // [256,128],[128]
    const float* dec_w; const float* dec_b;        // [128],[1]
    float* loss_acc;
    int Bg;                // num graphs
};

__global__ void mpnn_zero_kernel(float* p) {
    if (threadIdx.x == 0 && blockIdx.x == 0) *p = 0.0f;
}

__global__ void mpnn_finalize_kernel(const float* acc, float* out, float inv) {
    if (threadIdx.x == 0 && blockIdx.x == 0) out[0] = acc[0] * inv;
}

__global__ __launch_bounds__(256) void mpnn_fused_kernel(MpnnParams p) {
    __shared__ _Float16 s_nf[NODES][DM];   //  5 KB  node features (f16)
    __shared__ _Float16 s_ef[EDGES][DM];   // 25 KB  edge features (f16)
    __shared__ float    s_agg[NODES][DM];  // 10 KB  aggregation (f32) / weight staging
    __shared__ float    s_x[NODES][16];    //  1.25 KB
    __shared__ float    s_ea[EDGES][16];   //  6.25 KB
    __shared__ int      s_src[EDGES];
    __shared__ int      s_dst[EDGES];
    __shared__ float    s_hid[NODES];
    __shared__ float    s_loss;

    const int t    = threadIdx.x;
    const int lane = t & 31;
    const int wave = t >> 5;
    const int g0       = blockIdx.x * GPB;
    const int nodeBase = g0 * NPG;
    const int edgeBase = g0 * EPG;
    const int Etot     = p.Bg * EPG;

    float* aggF = &s_agg[0][0];            // flat view (2560 floats >= 2048 for staging)

    // ---- Phase 0: stage raw inputs + node-encoder weight ----
#if HAVE_ASYNC_LDS
    {
        // Stream x / edge_attr rows memory->LDS with no VGPR staging (ASYNCcnt path).
        const char* gx = (const char*)(p.x + nodeBase * 16);
        char*       lx = (char*)&s_x[0][0];
        for (int i = t; i < (NODES * 16) / 4; i += 256)   // 80 x 16B
            __builtin_amdgcn_global_load_async_to_lds_b128(
                (vi4*)(gx + 16 * i), (vi4*)(lx + 16 * i), 0, 0);
        const char* ge = (const char*)(p.ea + edgeBase * 16);
        char*       le = (char*)&s_ea[0][0];
        for (int i = t; i < (EDGES * 16) / 4; i += 256)   // 400 x 16B
            __builtin_amdgcn_global_load_async_to_lds_b128(
                (vi4*)(ge + 16 * i), (vi4*)(le + 16 * i), 0, 0);
    }
#else
    for (int i = t; i < NODES * 16; i += 256) s_x[i >> 4][i & 15] = p.x[nodeBase * 16 + i];
    for (int i = t; i < EDGES * 16; i += 256) s_ea[i >> 4][i & 15] = p.ea[edgeBase * 16 + i];
#endif
    for (int e = t; e < EDGES; e += 256) {
        s_src[e] = p.ei[edgeBase + e] - nodeBase;          // local src in [0,NODES)
        s_dst[e] = p.ei[Etot + edgeBase + e] - nodeBase;   // local dst in [0,NODES)
    }
    for (int i = t; i < 16 * DM; i += 256) aggF[i] = p.ne_w[i];
    if (t == 0) s_loss = 0.0f;
#if HAVE_ASYNC_LDS
    __builtin_amdgcn_s_wait_asynccnt(0);   // drain ASYNCcnt before waves consume LDS
#endif
    __syncthreads();

    // ---- Phase 1: node encoder  nf = x @ Wne + b  ->  f16 LDS ----
    for (int o = t; o < NODES * DM; o += 256) {
        int n = o >> 7, c = o & 127;
        float acc = p.ne_b[c];
#pragma unroll
        for (int k = 0; k < 16; ++k) acc += s_x[n][k] * aggF[k * DM + c];
        s_nf[n][c] = (_Float16)acc;
    }
    __syncthreads();
    for (int i = t; i < 16 * DM; i += 256) aggF[i] = p.ee_w[i];   // restage edge-enc weight
    __syncthreads();

    // ---- Phase 2: edge encoder  ef = ea @ Wee + b  ->  f16 LDS ----
    for (int o = t; o < EDGES * DM; o += 256) {
        int e = o >> 7, c = o & 127;
        float acc = p.ee_b[c];
#pragma unroll
        for (int k = 0; k < 16; ++k) acc += s_ea[e][k] * aggF[k * DM + c];
        s_ef[e][c] = (_Float16)acc;
    }
    __syncthreads();

    // Per-wave output-channel column and K sub-offset per WMMA 16-bit layout:
    //   lane < 16 : row M = lane,    K in {0..7, 16..23}
    //   lane >= 16: row M = lane-16, K in {8..15, 24..31}
    const int nCol = (wave << 4) + (lane & 15);   // this wave's 16 channels: [16w,16w+16)
    const int kHi  = (lane >> 4) << 3;            // 0 or 8

    v16h bf[KSTEPS];                              // register-resident B fragments

    for (int l = 0; l < 4; ++l) {
        // zero aggregation buffer
        for (int i = t; i < NODES * DM; i += 256) aggF[i] = 0.0f;

        // warm L2/L0 for the next layer's weights (global_prefetch_b8)
        if (l < 3) {
            __builtin_prefetch(p.msg_w[l + 1] + t, 0, 1);
            __builtin_prefetch(p.upd_w[l + 1] + t, 0, 1);
        }

        // ---- load message-weight B fragments: W[k][n], f32 -> f16 ----
        {
            const float* W = p.msg_w[l];
#pragma unroll
            for (int kt = 0; kt < KSTEPS; ++kt) {
                int kb = kt * 32 + kHi;
                v16h f;
#pragma unroll
                for (int j = 0; j < 8; ++j) {
                    f[j]     = (_Float16)W[(kb + j) * DM + nCol];
                    f[j + 8] = (_Float16)W[(kb + 16 + j) * DM + nCol];
                }
                bf[kt] = f;
            }
        }
        const float mb = p.msg_b[l][nCol];
        __syncthreads();   // agg zeroing complete before atomics

        // ---- message GEMM:  relu( concat(nf[src], ef) @ Wmsg + b ), scatter-add to agg ----
        for (int mt = 0; mt < ETILES; ++mt) {
            const int r = (mt << 4) + (lane & 15);     // edge row for this lane's A frag
            v8f acc = {};
#pragma unroll
            for (int kt = 0; kt < KSTEPS; ++kt) {
                v16h a = {};
                if (r < EDGES) {
                    const int kb = kt * 32 + kHi;
                    const _Float16* row = (kt < 4) ? &s_nf[s_src[r]][0]
                                                   : (&s_ef[r][0] - 128);
#pragma unroll
                    for (int j = 0; j < 8; ++j) {
                        a[j]     = row[kb + j];
                        a[j + 8] = row[kb + 16 + j];
                    }
                }
                acc = __builtin_amdgcn_wmma_f32_16x16x32_f16(
                          false, a, false, bf[kt], (short)0, acc, false, false);
            }
            // C layout: element i -> edge M = 16*mt + i + 8*(lane>=16), channel = nCol
#pragma unroll
            for (int i = 0; i < 8; ++i) {
                const int e = (mt << 4) + i + ((lane >> 4) << 3);
                if (e < EDGES) {
                    float v = acc[i] + mb;
                    v = v > 0.0f ? v : 0.0f;
                    atomicAdd(&s_agg[s_dst[e]][nCol], v);   // ds_add_f32
                }
            }
        }
        __syncthreads();   // all messages aggregated

        // ---- load update-weight B fragments ----
        {
            const float* W = p.upd_w[l];
#pragma unroll
            for (int kt = 0; kt < KSTEPS; ++kt) {
                int kb = kt * 32 + kHi;
                v16h f;
#pragma unroll
                for (int j = 0; j < 8; ++j) {
                    f[j]     = (_Float16)W[(kb + j) * DM + nCol];
                    f[j + 8] = (_Float16)W[(kb + 16 + j) * DM + nCol];
                }
                bf[kt] = f;
            }
        }
        const float ub = p.upd_b[l][nCol];

        // ---- update GEMM: relu( concat(nf, agg) @ Wupd + b ) into registers ----
        v8f uacc[NTILES];
        for (int mt = 0; mt < NTILES; ++mt) {
            const int r = (mt << 4) + (lane & 15);     // node row
            v8f acc = {};
#pragma unroll
            for (int kt = 0; kt < KSTEPS; ++kt) {
                v16h a = {};
                if (r < NODES) {
                    const int kb = kt * 32 + kHi;
                    if (kt < 4) {
                        const _Float16* row = &s_nf[r][0];
#pragma unroll
                        for (int j = 0; j < 8; ++j) {
                            a[j]     = row[kb + j];
                            a[j + 8] = row[kb + 16 + j];
                        }
                    } else {
                        const float* row = &s_agg[r][0] - 128;
#pragma unroll
                        for (int j = 0; j < 8; ++j) {
                            a[j]     = (_Float16)row[kb + j];
                            a[j + 8] = (_Float16)row[kb + 16 + j];
                        }
                    }
                }
                acc = __builtin_amdgcn_wmma_f32_16x16x32_f16(
                          false, a, false, bf[kt], (short)0, acc, false, false);
            }
            uacc[mt] = acc;
        }
        __syncthreads();   // everyone done reading s_nf / s_agg

        // ---- residual writeback: nf += relu(upd) ----
        for (int mt = 0; mt < NTILES; ++mt) {
#pragma unroll
            for (int i = 0; i < 8; ++i) {
                const int n = (mt << 4) + i + ((lane >> 4) << 3);
                if (n < NODES) {
                    float v = uacc[mt][i] + ub;
                    v = v > 0.0f ? v : 0.0f;
                    s_nf[n][nCol] = (_Float16)((float)s_nf[n][nCol] + v);
                }
            }
        }
        __syncthreads();
    }

    // ---- decoder + loss ----
    for (int n = t; n < NODES; n += 256) {
        float acc = p.dec_b[0];
        for (int c = 0; c < DM; ++c) acc += (float)s_nf[n][c] * p.dec_w[c];
        s_hid[n] = acc;
    }
    __syncthreads();
    if (t < GPB * 5) {
        const int g = t / 5, s = t % 5;
        const float pred = s_hid[g * NPG + s] * s_hid[g * NPG + 5 + s];
        const float yv   = p.y[(g0 + g) * 5 + s];
        const float d    = pred - yv;
        atomicAdd(&s_loss, d * d);
    }
    __syncthreads();
    if (t == 0) atomicAdd(p.loss_acc, s_loss);
}

extern "C" void kernel_launch(void* const* d_in, const int* in_sizes, int n_in,
                              void* d_out, int out_size, void* d_ws, size_t ws_size,
                              hipStream_t stream) {
    MpnnParams p;
    p.x    = (const float*)d_in[0];
    p.ei   = (const int*)  d_in[1];
    p.ea   = (const float*)d_in[2];
    p.y    = (const float*)d_in[3];
    p.ne_w = (const float*)d_in[4];  p.ne_b = (const float*)d_in[5];
    p.ee_w = (const float*)d_in[6];  p.ee_b = (const float*)d_in[7];
    for (int l = 0; l < 4; ++l) {
        p.msg_w[l] = (const float*)d_in[8 + 4 * l];
        p.msg_b[l] = (const float*)d_in[9 + 4 * l];
        p.upd_w[l] = (const float*)d_in[10 + 4 * l];
        p.upd_b[l] = (const float*)d_in[11 + 4 * l];
    }
    p.dec_w = (const float*)d_in[24];
    p.dec_b = (const float*)d_in[25];

    const int Bg = in_sizes[3] / 5;     // y has B*5 elements
    p.Bg = Bg;
    p.loss_acc = (float*)d_ws;

    float* out = (float*)d_out;
    const float inv = 1.0f / (float)(Bg * 5);

    mpnn_zero_kernel<<<1, 64, 0, stream>>>(p.loss_acc);
    mpnn_fused_kernel<<<Bg / GPB, 256, 0, stream>>>(p);
    mpnn_finalize_kernel<<<1, 64, 0, stream>>>(p.loss_acc, out, inv);
}
